// VectorQuantizer_14508399526337
// MI455X (gfx1250) — compile-verified, matrix operand order checked by reading, not run on hardware
//
#include <hip/hip_runtime.h>
#include <hip/hip_bf16.h>

#define K_DIM   1024
#define N_ROWS  131072
#define TILE_M  16

typedef _Float16 v16h __attribute__((ext_vector_type(16)));
typedef _Float16 v8h  __attribute__((ext_vector_type(8)));
typedef float    v8f  __attribute__((ext_vector_type(8)));
typedef float    v4f  __attribute__((ext_vector_type(4)));

// Map float to an unsigned so that unsigned compare == float compare.
__device__ __forceinline__ unsigned orderable_u32(float f) {
    unsigned u = __float_as_uint(f);
    return (u & 0x80000000u) ? ~u : (u | 0x80000000u);
}

// ---------------------------------------------------------------------------
// Kernel 0: zero the argmax cell in workspace (harness does not re-poison).
// ---------------------------------------------------------------------------
__global__ void vq_init_kernel(unsigned long long* best) {
    if (threadIdx.x == 0 && blockIdx.x == 0) *best = 0ull;
}

// ---------------------------------------------------------------------------
// Kernel 1: streaming WMMA matvec + hierarchical argmax.
// One wave owns a 16-row tile of W; dots accumulated with
// v_wmma_f32_16x16x32_f16 (B = z broadcast across the N dimension).
// ---------------------------------------------------------------------------
__global__ void __launch_bounds__(256)
vq_argmax_kernel(const float* __restrict__ z,
                 const float* __restrict__ W,
                 unsigned long long* __restrict__ best) {
    __shared__ __align__(16) _Float16 zs[K_DIM];

    // Stage z in f16 into LDS once per block (2 KB).
    for (int i = threadIdx.x; i < K_DIM; i += blockDim.x)
        zs[i] = (_Float16)z[i];
    __syncthreads();

    const int lane  = threadIdx.x & 31;
    const int hgrp  = lane >> 4;      // which half of the wave (K-run selector)
    const int mrow  = lane & 15;      // row within the 16-row tile
    const int wave  = threadIdx.x >> 5;
    const int wavesPerBlock = blockDim.x >> 5;
    const int gwave = blockIdx.x * wavesPerBlock + wave;
    const int nwave = gridDim.x * wavesPerBlock;

    unsigned long long myBest = 0ull;

    for (int tile = gwave; tile < N_ROWS / TILE_M; tile += nwave) {
        const int rowbase = tile * TILE_M;
        const float* __restrict__ wrow = W + (size_t)(rowbase + mrow) * K_DIM;

        v8f c = {0.f, 0.f, 0.f, 0.f, 0.f, 0.f, 0.f, 0.f};

        #pragma unroll 2
        for (int kb = 0; kb < K_DIM; kb += 32) {
            const int k0 = kb + hgrp * 8;        // A VGPR0-3 K-run
            const int k1 = kb + 16 + hgrp * 8;   // A VGPR4-7 K-run

            // Prefetch this row ~512B ahead (global_prefetch_b8).
            __builtin_prefetch(wrow + k0 + 128, 0, 0);

            // W fragment: two 32B f32 chunks -> 4x global_load_b128 per lane.
            v4f wa0 = *(const v4f*)(wrow + k0);
            v4f wa1 = *(const v4f*)(wrow + k0 + 4);
            v4f wb0 = *(const v4f*)(wrow + k1);
            v4f wb1 = *(const v4f*)(wrow + k1 + 4);

            v16h a;
            #pragma unroll
            for (int i = 0; i < 4; ++i) {
                a[i]      = (_Float16)wa0[i];
                a[4 + i]  = (_Float16)wa1[i];
                a[8 + i]  = (_Float16)wb0[i];
                a[12 + i] = (_Float16)wb1[i];
            }

            // z fragment (broadcast across N): two 16B LDS loads (ds_load_b128).
            v8h bz0 = *(const v8h*)&zs[k0];
            v8h bz1 = *(const v8h*)&zs[k1];
            v16h b;
            #pragma unroll
            for (int i = 0; i < 8; ++i) { b[i] = bz0[i]; b[8 + i] = bz1[i]; }

            // D = A x B + C, f32 accumulate.
            c = __builtin_amdgcn_wmma_f32_16x16x32_f16(
                    /*neg_a=*/false, a, /*neg_b=*/false, b,
                    /*c_mod=*/(short)0, c, /*reuse_a=*/false, /*reuse_b=*/false);
        }

        // D layout: VGPR r holds row (r + 8*hgrp); all N columns identical.
        float bestv = -3.402823e38f;
        int   bestr = 0;
        #pragma unroll
        for (int r = 0; r < 8; ++r) {
            float v = c[r];
            if (v > bestv) { bestv = v; bestr = rowbase + r + 8 * hgrp; }
        }

        unsigned long long key =
            ((unsigned long long)orderable_u32(bestv) << 32) |
            (unsigned long long)(~(unsigned)bestr);   // ~idx: ties -> lowest index
        if (key > myBest) myBest = key;
    }

    // Wave-level butterfly max (wave32).
    #pragma unroll
    for (int off = 16; off > 0; off >>= 1) {
        unsigned long long o = __shfl_xor(myBest, off, 32);
        if (o > myBest) myBest = o;
    }
    if (lane == 0) atomicMax(best, myBest);
}

// ---------------------------------------------------------------------------
// Kernel 2: decode winner, emit quantized row, index, commitment loss.
// quantized_st = z + sg(q - z) == q numerically.
// ---------------------------------------------------------------------------
__global__ void __launch_bounds__(256)
vq_finalize_kernel(const float* __restrict__ z,
                   const float* __restrict__ W,
                   const unsigned long long* __restrict__ best,
                   float* __restrict__ out) {
    __shared__ float red[256];

    const unsigned long long key = *best;
    const int index = (int)(~(unsigned)(key & 0xFFFFFFFFull));
    const float* __restrict__ q = W + (size_t)index * K_DIM;

    float acc = 0.f;
    for (int i = threadIdx.x; i < K_DIM; i += blockDim.x) {
        float qi = q[i];
        out[i] = qi;
        float d = z[i] - qi;
        acc += d * d;
    }
    red[threadIdx.x] = acc;
    __syncthreads();
    for (int s = blockDim.x >> 1; s > 0; s >>= 1) {
        if ((int)threadIdx.x < s) red[threadIdx.x] += red[threadIdx.x + s];
        __syncthreads();
    }
    if (threadIdx.x == 0) {
        out[K_DIM]     = (float)index;                       // argmax index
        out[K_DIM + 1] = 0.25f * red[0] / (float)K_DIM;      // commitment loss
    }
}

// ---------------------------------------------------------------------------
extern "C" void kernel_launch(void* const* d_in, const int* in_sizes, int n_in,
                              void* d_out, int out_size, void* d_ws, size_t ws_size,
                              hipStream_t stream) {
    const float* z = (const float*)d_in[0];   // (1024,)  f32, unit norm
    const float* W = (const float*)d_in[1];   // (131072, 1024) f32, rows unit norm
    float* out = (float*)d_out;               // [1024 quantized | index | loss]
    unsigned long long* best = (unsigned long long*)d_ws;

    vq_init_kernel<<<1, 32, 0, stream>>>(best);

    // 8192 tiles of 16 rows; 1024 blocks x 8 waves -> one tile per wave.
    vq_argmax_kernel<<<1024, 256, 0, stream>>>(z, W, best);

    vq_finalize_kernel<<<1, 256, 0, stream>>>(z, W, best, out);
}
